// BaseRouter_24215025615336
// MI455X (gfx1250) — matrix operand order repaired and checked by reading, not compiled, not run on hardware
//
#include <hip/hip_runtime.h>
#include <hip/hip_bf16.h>
#include <math.h>

typedef __bf16 bf16;
typedef __attribute__((ext_vector_type(16))) __bf16 v16bf;
typedef __attribute__((ext_vector_type(8)))  __bf16 v8bf;
typedef __attribute__((ext_vector_type(8)))  float  v8f;

#define TOKENS 16384   // B*S
#define HDIM   2048
#define EDIM   2
#define LN_EPS 1e-5f
#define AUX_EPS 1e-9f

// ---------------- workspace layout (bytes) ----------------
#define XN_OFF   0UL          // Xn  bf16 [16384][2048] : 67108864 B
#define W1T_OFF  67108864UL   // W1t bf16 [2048][2048]  :  8388608 B
#define LOG_OFF  75497472UL   // logits f32 [16384][2]  :   131072 B
#define PSUM_OFF 75628544UL   // psum f32 [2]

// ---- CDNA5 async global->LDS copy (ASYNCcnt path, ISA §15.18.3 op 98) ----
__device__ __forceinline__ void async_b128(unsigned lds_off, const void* gptr) {
  asm volatile("global_load_async_to_lds_b128 %0, %1, off"
               :
               : "v"(lds_off), "v"((unsigned long long)(uintptr_t)gptr)
               : "memory");
}
__device__ __forceinline__ void wait_async0() {
  asm volatile("s_wait_asynccnt 0x0" ::: "memory");
}

// init logits with b2 (fused bias of 2nd linear), zero p_expert accumulators
__global__ void zero_kernel(float* __restrict__ logits, float* __restrict__ psum,
                            const float* __restrict__ b2) {
  int i = blockIdx.x * blockDim.x + threadIdx.x;
  if (i < TOKENS * EDIM) logits[i] = b2[i & 1];
  if (i < EDIM) psum[i] = 0.f;
}

// LayerNorm over H per token, write bf16
__global__ void __launch_bounds__(256) ln_kernel(const float* __restrict__ x,
                                                 const float* __restrict__ gamma,
                                                 const float* __restrict__ beta,
                                                 bf16* __restrict__ xn) {
  __shared__ float s1[256], s2[256];
  const int t = threadIdx.x;
  const size_t row = blockIdx.x;
  const float* xr = x + row * (size_t)HDIM;
  float v[8];
  float sum = 0.f, sq = 0.f;
#pragma unroll
  for (int i = 0; i < 8; ++i) {
    v[i] = xr[t + i * 256];
    sum += v[i];
    sq  += v[i] * v[i];
  }
  s1[t] = sum; s2[t] = sq;
  __syncthreads();
  for (int off = 128; off > 0; off >>= 1) {
    if (t < off) { s1[t] += s1[t + off]; s2[t] += s2[t + off]; }
    __syncthreads();
  }
  const float mu   = s1[0] * (1.0f / HDIM);
  const float var  = s2[0] * (1.0f / HDIM) - mu * mu;
  const float rstd = rsqrtf(var + LN_EPS);
  bf16* outr = xn + row * (size_t)HDIM;
#pragma unroll
  for (int i = 0; i < 8; ++i) {
    const int h = t + i * 256;
    const float y = (v[i] - mu) * rstd * gamma[h] + beta[h];
    outr[h] = (bf16)y;
  }
}

// transpose + f32->bf16 of W1: W1t[n][k] = W1[k][n]
__global__ void __launch_bounds__(256) w1t_kernel(const float* __restrict__ W1,
                                                  bf16* __restrict__ W1t) {
  __shared__ float tile[32][33];
  const int kBase = blockIdx.y * 32;
  const int nBase = blockIdx.x * 32;
  const int tx = threadIdx.x;         // 0..31
  const int ty = threadIdx.y;         // 0..7
#pragma unroll
  for (int i = 0; i < 32; i += 8)
    tile[ty + i][tx] = W1[(size_t)(kBase + ty + i) * HDIM + nBase + tx];
  __syncthreads();
#pragma unroll
  for (int i = 0; i < 32; i += 8)
    W1t[(size_t)(nBase + ty + i) * HDIM + kBase + tx] = (bf16)tile[tx][ty + i];
}

// 128x128 block tile GEMM (double-buffered LDS, async global->LDS):
// h = relu(Xn @ W1 + b1); fused partial logits += h @ W2
__global__ void __launch_bounds__(256)
gemm_router_kernel(const bf16* __restrict__ Xn, const bf16* __restrict__ W1t,
                   const float* __restrict__ bias1, const float* __restrict__ W2,
                   float* __restrict__ logits) {
  __shared__ __align__(32) bf16 As[2][128][64];   // [buf][m][k]  16 KB each
  __shared__ __align__(32) bf16 Bs[2][128][64];   // [buf][n][k]  (W1 transposed)
  __shared__ float ldsLog[128][2];

  const int t    = threadIdx.x;
  const int w    = t >> 5;       // wave id 0..7
  const int lane = t & 31;
  const int half = lane >> 4;
  const int nl   = lane & 15;

  const int mBase = blockIdx.y * 128;
  const int nBase = blockIdx.x * 128;
  const int mrowB = (w & 3) * 32;   // wave row base in block tile (2 m-tiles)
  const int ncolB = (w >> 2) * 64;  // wave col base in block tile (4 n-tiles)

  ldsLog[t >> 1][t & 1] = 0.f;

  // ---- async staging setup: thread t covers row t>>1, k half sel = t&1 ----
  const int arow = t >> 1;            // 0..127
  const int ksel = (t & 1) * 32;      // 0 or 32
  const unsigned thrOff = (unsigned)((arow * 64 + ksel) * 2);
  const unsigned ldsA = (unsigned)(uintptr_t)&As[0][0][0] + thrOff;
  const unsigned ldsB = (unsigned)(uintptr_t)&Bs[0][0][0] + thrOff;
  const bf16* aG = Xn  + (size_t)(mBase + arow) * HDIM + ksel;
  const bf16* bG = W1t + (size_t)(nBase + arow) * HDIM + ksel;

  auto stage = [&](int st, int buf) {
    const int kt = st * 64;
    const unsigned aL = ldsA + (unsigned)buf * 16384u;
    const unsigned bL = ldsB + (unsigned)buf * 16384u;
#pragma unroll
    for (int j = 0; j < 4; ++j) {            // 4 x 16B chunks = 32 bf16
      async_b128(aL + j * 16u, aG + kt + j * 8);
      async_b128(bL + j * 16u, bG + kt + j * 8);
    }
  };

  v8f acc[2][4] = {};   // [m-tile][n-tile]

  stage(0, 0);
  wait_async0();

  const int NSTAGE = HDIM / 64;   // 32
  for (int st = 0; st < NSTAGE; ++st) {
    const int c = st & 1;
    __syncthreads();                       // buffer c ready for all waves
    if (st + 1 < NSTAGE) stage(st + 1, c ^ 1);   // prefetch overlaps WMMAs

#pragma unroll
    for (int sub = 0; sub < 2; ++sub) {
      const int ks = sub * 32;
      // A fragments: lane = row M; K = 16*half + e (contiguous)
      v16bf af[2];
#pragma unroll
      for (int mi = 0; mi < 2; ++mi) {
        const int row = mrowB + mi * 16 + nl;
        const v8bf alo = *(const v8bf*)&As[c][row][ks + 16 * half];
        const v8bf ahi = *(const v8bf*)&As[c][row][ks + 16 * half + 8];
#pragma unroll
        for (int e = 0; e < 8; ++e) { af[mi][e] = alo[e]; af[mi][e + 8] = ahi[e]; }
      }
      // B fragments: lane = col N; K = 16*half + e (contiguous)
#pragma unroll
      for (int nj = 0; nj < 4; ++nj) {
        const int col = ncolB + nj * 16 + nl;
        const v16bf bb = *(const v16bf*)&Bs[c][col][ks + 16 * half];
#pragma unroll
        for (int mi = 0; mi < 2; ++mi)
          acc[mi][nj] = __builtin_amdgcn_wmma_f32_16x16x32_bf16(
              false, af[mi], false, bb, (short)0, acc[mi][nj], false, false);
      }
    }
    wait_async0();   // own prefetch complete before next barrier
  }

  // epilogue: bias + ReLU + h@W2 (E=2), reduce over this wave's 4 n-tiles first
  {
    float bi[4], wa[4], wb[4];
#pragma unroll
    for (int nj = 0; nj < 4; ++nj) {
      const int gcol = nBase + ncolB + nj * 16 + nl;
      bi[nj] = bias1[gcol];
      wa[nj] = W2[gcol * 2 + 0];
      wb[nj] = W2[gcol * 2 + 1];
    }
#pragma unroll
    for (int mi = 0; mi < 2; ++mi) {
#pragma unroll
      for (int r = 0; r < 8; ++r) {
        float c0 = 0.f, c1 = 0.f;
#pragma unroll
        for (int nj = 0; nj < 4; ++nj) {
          float h = acc[mi][nj][r] + bi[nj];
          h = h > 0.f ? h : 0.f;
          c0 += h * wa[nj];
          c1 += h * wb[nj];
        }
        const int br = mrowB + mi * 16 + r + 8 * half;  // C layout: M = r + 8*half
        atomicAdd(&ldsLog[br][0], c0);
        atomicAdd(&ldsLog[br][1], c1);
      }
    }
  }
  __syncthreads();
  atomicAdd(&logits[(size_t)(mBase + (t >> 1)) * 2 + (t & 1)], ldsLog[t >> 1][t & 1]);
}

// softmax (E=2) + top-2 + renorm + p_expert partial sums
__global__ void __launch_bounds__(256)
softmax_topk_kernel(const float* __restrict__ logits, float* __restrict__ out,
                    float* __restrict__ psum) {
  __shared__ float s0[256], s1[256];
  const int tk = blockIdx.x * 256 + threadIdx.x;
  const float l0 = logits[tk * 2 + 0];
  const float l1 = logits[tk * 2 + 1];
  const float m  = fmaxf(l0, l1);
  const float e0 = expf(l0 - m), e1 = expf(l1 - m);
  const float inv = 1.0f / (e0 + e1);
  const float p0 = e0 * inv, p1 = e1 * inv;
  float t0, t1, i0, i1;
  if (p0 >= p1) { t0 = p0; t1 = p1; i0 = 0.f; i1 = 1.f; }  // tie -> lower index first
  else          { t0 = p1; t1 = p0; i0 = 1.f; i1 = 0.f; }
  const float sinv = 1.0f / (t0 + t1);
  out[tk * 2 + 0] = i0;
  out[tk * 2 + 1] = i1;
  out[TOKENS * 2 + tk * 2 + 0] = t0 * sinv;
  out[TOKENS * 2 + tk * 2 + 1] = t1 * sinv;

  s0[threadIdx.x] = p0; s1[threadIdx.x] = p1;
  __syncthreads();
  for (int off = 128; off > 0; off >>= 1) {
    if (threadIdx.x < off) {
      s0[threadIdx.x] += s0[threadIdx.x + off];
      s1[threadIdx.x] += s1[threadIdx.x + off];
    }
    __syncthreads();
  }
  if (threadIdx.x == 0) {
    atomicAdd(&psum[0], s0[0]);
    atomicAdd(&psum[1], s1[0]);
  }
}

__global__ void aux_kernel(const float* __restrict__ psum, float* __restrict__ out) {
  if (threadIdx.x == 0 && blockIdx.x == 0) {
    float aux = 0.f;
#pragma unroll
    for (int e = 0; e < EDIM; ++e) {
      const float pe = psum[e] * (1.0f / TOKENS);
      aux += pe * logf(pe * EDIM + AUX_EPS);
    }
    out[TOKENS * 4] = aux;  // after 2*K*TOKENS index + prob elements
  }
}

extern "C" void kernel_launch(void* const* d_in, const int* in_sizes, int n_in,
                              void* d_out, int out_size, void* d_ws, size_t ws_size,
                              hipStream_t stream) {
  const float* x     = (const float*)d_in[0];
  const float* W1    = (const float*)d_in[1];
  const float* b1    = (const float*)d_in[2];
  const float* W2    = (const float*)d_in[3];
  const float* b2    = (const float*)d_in[4];
  const float* gamma = (const float*)d_in[5];
  const float* beta  = (const float*)d_in[6];
  float* out = (float*)d_out;

  char* ws = (char*)d_ws;
  bf16*  Xn     = (bf16*)(ws + XN_OFF);
  bf16*  W1t    = (bf16*)(ws + W1T_OFF);
  float* logits = (float*)(ws + LOG_OFF);
  float* psum   = (float*)(ws + PSUM_OFF);

  zero_kernel<<<(TOKENS * EDIM + 255) / 256, 256, 0, stream>>>(logits, psum, b2);
  ln_kernel<<<TOKENS, 256, 0, stream>>>(x, gamma, beta, Xn);
  w1t_kernel<<<dim3(HDIM / 32, HDIM / 32), dim3(32, 8), 0, stream>>>(W1, W1t);
  gemm_router_kernel<<<dim3(HDIM / 128, TOKENS / 128), 256, 0, stream>>>(Xn, W1t, b1, W2, logits);
  softmax_topk_kernel<<<TOKENS / 256, 256, 0, stream>>>(logits, out, psum);
  aux_kernel<<<1, 1, 0, stream>>>(psum, out);
}
